// DendriticCompartment_17910013624698
// MI455X (gfx1250) — matrix operand order repaired and checked by reading, not compiled, not run on hardware
//
#include <hip/hip_runtime.h>
#include <hip/hip_bf16.h>
#include <math.h>

#define B_ROWS 8192
#define D_IN   2048
#define D_ST   2048
#define N_SEG  4

#define BLK_M 128
#define BLK_N 128
#define BLK_K 32
#define LDSS  40   // padded LDS row stride in bf16 elems (80 B) -> conflict-free ds_load_b128

typedef __attribute__((ext_vector_type(16))) __bf16 v16bf;
typedef __attribute__((ext_vector_type(8)))  __bf16 v8bf;
typedef __attribute__((ext_vector_type(4)))  __bf16 v4bf;
typedef __attribute__((ext_vector_type(8)))  float  v8f;

__device__ __forceinline__ v16bf cat8(v8bf a, v8bf b) {
  return __builtin_shufflevector(a, b, 0,1,2,3,4,5,6,7,8,9,10,11,12,13,14,15);
}

struct bfpair { __bf16 h, l; };
__device__ __forceinline__ bfpair split_bf16(float f) {
  bfpair p;
  p.h = (__bf16)f;
  p.l = (__bf16)(f - (float)p.h);
  return p;
}

// ---------------------------------------------------------------------------
// Kernel 1: gates[b][s] = sigmoid(x[b,:] . W_gate[s,:] + b_gate[s])
// One wave per row; 4 dot products per wave; shuffle-xor reduction (wave32).
// ---------------------------------------------------------------------------
__global__ __launch_bounds__(256)
void dendrite_gates_kernel(const float* __restrict__ x,
                           const float* __restrict__ Wg,
                           const float* __restrict__ bg,
                           float* __restrict__ gates)
{
  const int lane = threadIdx.x & 31;
  const int row  = blockIdx.x * (blockDim.x >> 5) + (threadIdx.x >> 5);
  if (row >= B_ROWS) return;

  const float* xr = x + (size_t)row * D_IN;
  float a0 = 0.f, a1 = 0.f, a2 = 0.f, a3 = 0.f;
  for (int i = lane; i < D_IN; i += 32) {
    const float xv = xr[i];
    a0 = fmaf(xv, Wg[0 * D_IN + i], a0);
    a1 = fmaf(xv, Wg[1 * D_IN + i], a1);
    a2 = fmaf(xv, Wg[2 * D_IN + i], a2);
    a3 = fmaf(xv, Wg[3 * D_IN + i], a3);
  }
#pragma unroll
  for (int off = 16; off > 0; off >>= 1) {
    a0 += __shfl_xor(a0, off, 32);
    a1 += __shfl_xor(a1, off, 32);
    a2 += __shfl_xor(a2, off, 32);
    a3 += __shfl_xor(a3, off, 32);
  }
  if (lane < 4) {
    float a = (lane == 0) ? a0 : (lane == 1) ? a1 : (lane == 2) ? a2 : a3;
    a += bg[lane];
    gates[(size_t)row * N_SEG + lane] = 1.0f / (1.0f + __expf(-a));
  }
}

// ---------------------------------------------------------------------------
// Kernel 2: per 128x128 output tile, loop segments; split-bf16 WMMA GEMM with
// f32 accumulation; running sum & signed product across segments in registers;
// fused sigmoid-plateau / gate / root-mean epilogue.
// ---------------------------------------------------------------------------
__global__ __launch_bounds__(256)
void dendrite_main_kernel(const float* __restrict__ x,
                          const float* __restrict__ Wseg,
                          const float* __restrict__ bseg,
                          const float* __restrict__ thr,
                          const float* __restrict__ gates,
                          float* __restrict__ out)
{
  __shared__ __bf16 sXhi[BLK_M * LDSS];
  __shared__ __bf16 sXlo[BLK_M * LDSS];
  __shared__ __bf16 sWhi[BLK_N * LDSS];
  __shared__ __bf16 sWlo[BLK_N * LDSS];

  const int t    = threadIdx.x;
  const int lane = t & 31;
  const int wv   = t >> 5;       // wave 0..7
  const int wm   = wv & 1;       // row half of C block (64 rows)
  const int wn   = wv >> 1;      // col quarter of C block (32 cols)
  const int l15  = lane & 15;
  const int kh   = lane >> 4;    // lane-half selects K half (16 contiguous)

  const int rowBlk = blockIdx.y * BLK_M;
  const int colBlk = blockIdx.x * BLK_N;

  // staging: 256 threads fill 128x32 f32 tiles, float4 per thread, 4 row-passes
  const int lr = t >> 3;          // 0..31
  const int lc = (t & 7) * 4;     // 0..28

  const v8f vzero = {};
  const v8f vone  = vzero + 1.0f;

  v8f ssum[4][2], sprd[4][2];
#pragma unroll
  for (int m = 0; m < 4; ++m)
#pragma unroll
    for (int n = 0; n < 2; ++n) { ssum[m][n] = vzero; sprd[m][n] = vone; }

  for (int s = 0; s < N_SEG; ++s) {
    const float* Ws = Wseg + (size_t)s * D_ST * D_IN;

    v8f acc[4][2];
#pragma unroll
    for (int m = 0; m < 4; ++m)
#pragma unroll
      for (int n = 0; n < 2; ++n) acc[m][n] = vzero;

    for (int k0 = 0; k0 < D_IN; k0 += BLK_K) {
      __syncthreads();   // previous iteration's fragment reads done
#pragma unroll
      for (int rr = 0; rr < BLK_M; rr += 32) {
        const int r = rr + lr;
        const float4 xv  = *(const float4*)(x  + (size_t)(rowBlk + r) * D_IN + k0 + lc);
        const float4 wvv = *(const float4*)(Ws + (size_t)(colBlk + r) * D_IN + k0 + lc);
        v4bf xh, xl, wh, wl;
        bfpair p;
        p = split_bf16(xv.x);  xh[0] = p.h; xl[0] = p.l;
        p = split_bf16(xv.y);  xh[1] = p.h; xl[1] = p.l;
        p = split_bf16(xv.z);  xh[2] = p.h; xl[2] = p.l;
        p = split_bf16(xv.w);  xh[3] = p.h; xl[3] = p.l;
        p = split_bf16(wvv.x); wh[0] = p.h; wl[0] = p.l;
        p = split_bf16(wvv.y); wh[1] = p.h; wl[1] = p.l;
        p = split_bf16(wvv.z); wh[2] = p.h; wl[2] = p.l;
        p = split_bf16(wvv.w); wh[3] = p.h; wl[3] = p.l;
        *(v4bf*)&sXhi[r * LDSS + lc] = xh;
        *(v4bf*)&sXlo[r * LDSS + lc] = xl;
        *(v4bf*)&sWhi[r * LDSS + lc] = wh;
        *(v4bf*)&sWlo[r * LDSS + lc] = wl;
      }
      __syncthreads();

      // B fragments (W side): lane n holds col=base+n; K contiguous 16 per lane-half
      v16bf bh[2], bl[2];
#pragma unroll
      for (int n = 0; n < 2; ++n) {
        const int col = wn * 32 + n * 16 + l15;
        const __bf16* ph = &sWhi[col * LDSS + kh * 16];
        const __bf16* pl = &sWlo[col * LDSS + kh * 16];
        bh[n] = cat8(*(const v8bf*)ph, *(const v8bf*)(ph + 8));
        bl[n] = cat8(*(const v8bf*)pl, *(const v8bf*)(pl + 8));
      }

#pragma unroll
      for (int m = 0; m < 4; ++m) {
        const int row = wm * 64 + m * 16 + l15;
        const __bf16* pah = &sXhi[row * LDSS];
        const __bf16* pal = &sXlo[row * LDSS];
        // A layout: lane-half kh holds K {16kh..16kh+15} contiguous
        const v16bf ah = cat8(*(const v8bf*)(pah + kh * 16), *(const v8bf*)(pah + kh * 16 + 8));
        const v16bf al = cat8(*(const v8bf*)(pal + kh * 16), *(const v8bf*)(pal + kh * 16 + 8));
#pragma unroll
        for (int n = 0; n < 2; ++n) {
          acc[m][n] = __builtin_amdgcn_wmma_f32_16x16x32_bf16(
              false, al, false, bh[n], (short)0, acc[m][n], false, false);
          acc[m][n] = __builtin_amdgcn_wmma_f32_16x16x32_bf16(
              false, ah, false, bl[n], (short)0, acc[m][n], false, false);
          acc[m][n] = __builtin_amdgcn_wmma_f32_16x16x32_bf16(
              false, ah, false, bh[n], (short)0, acc[m][n], false, false);
        }
      }
    }

    // Segment epilogue: bias + plateau + gate; fold into running sum / product.
#pragma unroll
    for (int m = 0; m < 4; ++m) {
      const int rbase = rowBlk + wm * 64 + m * 16 + kh * 8;
#pragma unroll
      for (int n = 0; n < 2; ++n) {
        const int c   = colBlk + wn * 32 + n * 16 + l15;
        const float bsv = bseg[(size_t)s * D_ST + c];
        const float thv = thr [(size_t)s * D_ST + c];
#pragma unroll
        for (int j = 0; j < 8; ++j) {
          const float v  = acc[m][n][j] + bsv;
          const float pl = 1.0f / (1.0f + __expf(-5.0f * (v - thv)));
          const float st = v * pl * gates[(size_t)(rbase + j) * N_SEG + s];
          ssum[m][n][j] += st;
          sprd[m][n][j] *= st;
        }
      }
    }
  }

  // integrated = sum + 0.1 * sign(prod) * |prod|^(1/4)
#pragma unroll
  for (int m = 0; m < 4; ++m) {
    const int rbase = rowBlk + wm * 64 + m * 16 + kh * 8;
#pragma unroll
    for (int n = 0; n < 2; ++n) {
      const int c = colBlk + wn * 32 + n * 16 + l15;
#pragma unroll
      for (int j = 0; j < 8; ++j) {
        const float pv = sprd[m][n][j];
        const float sg = (pv > 0.f) ? 1.f : ((pv < 0.f) ? -1.f : 0.f);
        out[(size_t)(rbase + j) * D_ST + c] =
            ssum[m][n][j] + 0.1f * sg * powf(fabsf(pv), 0.25f);
      }
    }
  }
}

// ---------------------------------------------------------------------------
extern "C" void kernel_launch(void* const* d_in, const int* in_sizes, int n_in,
                              void* d_out, int out_size, void* d_ws, size_t ws_size,
                              hipStream_t stream) {
  const float* x    = (const float*)d_in[0];
  const float* Wseg = (const float*)d_in[1];
  const float* bseg = (const float*)d_in[2];
  const float* thr  = (const float*)d_in[3];
  const float* Wg   = (const float*)d_in[4];
  const float* bg   = (const float*)d_in[5];
  float* out   = (float*)d_out;
  float* gates = (float*)d_ws;   // B_ROWS * N_SEG floats = 128 KB scratch

  dendrite_gates_kernel<<<dim3(B_ROWS / 8), 256, 0, stream>>>(x, Wg, bg, gates);
  dendrite_main_kernel<<<dim3(D_ST / BLK_N, B_ROWS / BLK_M), 256, 0, stream>>>(
      x, Wseg, bseg, thr, gates, out);
}
